// NLGNN_44959717655299
// MI455X (gfx1250) — matrix operand order repaired and checked by reading, not compiled, not run on hardware
//
#include <hip/hip_runtime.h>
#include <math.h>

// ---------------------------------------------------------------------------
// NLGNN pipeline for gfx1250 (MI455X): f16 WMMA GEMMs/convs, f32 accumulate.
// B matrices pre-swizzled to fragment-major layout for 128-bit loads.
// ---------------------------------------------------------------------------

#define F_IN 512
#define H1   256
#define C_OUT 40
#define KW   5
#define PADW 2

typedef __attribute__((ext_vector_type(16))) _Float16 v16h;
typedef __attribute__((ext_vector_type(8)))  _Float16 v8h;
typedef __attribute__((ext_vector_type(2)))  _Float16 v2h;
typedef __attribute__((ext_vector_type(8)))  float    v8f;

// ---- WMMA A fragment from LDS (16-bit A 16x32, contiguous k order) -----
// lane L: M = L&15, half = L>>4; VGPR v in 0..7 holds K = half*16+2v(+1)
// (16 contiguous halves, matching packed B) -> compiler emits 2x ds_load_b128.
__device__ __forceinline__ v16h load_afrag_lds(const _Float16* lds, int stride,
                                               int row_off, int kc) {
  int lane = threadIdx.x & 31;
  int m = lane & 15;
  int h = lane >> 4;
  v16h a;
#pragma unroll
  for (int v = 0; v < 8; ++v) {
    int kk = h * 16 + 2 * v;
    const _Float16* p = lds + (size_t)(row_off + m) * stride + kc + kk;
    a[2 * v]     = p[0];
    a[2 * v + 1] = p[1];
  }
  return a;
}

// ---- weight prep: pack f32 [Kdim x NC] matrix into fragment-major f16 ------
// dst half index: ((colTile*(Kdim/32)+kcIdx)*32 + lane)*16 + 2v + lohi
// holding B element (K = kcIdx*32 + (lane>>4)*16 + 2v + lohi, col = colTile*16 + (lane&15))
__global__ void pack_b(const float* __restrict__ src, _Float16* __restrict__ dst,
                       int Kdim, int NC) {
  int t = blockIdx.x * 256 + threadIdx.x;
  if (t >= Kdim * NC) return;
  int f = t >> 9, w = t & 511;
  int lane = w >> 4, ev = w & 15;
  int v = ev >> 1, lohi = ev & 1;
  int nK = Kdim >> 5;
  int colTile = f / nK, kcIdx = f - colTile * nK;
  int n = lane & 15, kb = (lane >> 4) * 16;
  int K = kcIdx * 32 + kb + 2 * v + lohi;
  int col = colTile * 16 + n;
  dst[t] = (_Float16)src[(size_t)K * NC + col];
}

// conv weights (co,ci,k) f32 -> per-k fragment-major f16 (B = W[k][ci][co])
__global__ void pack_convw(const float* __restrict__ src, _Float16* __restrict__ dst) {
  const int per_k = H1 * H1;
  int t = blockIdx.x * 256 + threadIdx.x;
  if (t >= KW * per_k) return;
  int k = t / per_k, u = t - k * per_k;
  int f = u >> 9, w = u & 511;
  int lane = w >> 4, ev = w & 15;
  int v = ev >> 1, lohi = ev & 1;
  const int nK = H1 >> 5;
  int colTile = f / nK, kcIdx = f - colTile * nK;
  int n = lane & 15, kb = (lane >> 4) * 16;
  int ci = kcIdx * 32 + kb + 2 * v + lohi;
  int co = colTile * 16 + n;
  dst[t] = (_Float16)src[(size_t)co * (H1 * KW) + (size_t)ci * KW + k];
}

// ---- GEMM 1: h0 = relu(x @ W1 + b1), x f32 [N,512] -> h0 f16 [N,256] -------

__global__ void gemm1_relu(const float* __restrict__ X,
                           const _Float16* __restrict__ W1p,
                           const float* __restrict__ b1,
                           _Float16* __restrict__ H0, int N) {
  __shared__ _Float16 ldsA[16 * 520];            // 16 rows x 512 K (stride 520)
  int rows0 = blockIdx.x * 16;
  int wave  = threadIdx.x >> 5;
  int lane  = threadIdx.x & 31;
  int colTile = blockIdx.y * 8 + wave;           // 16 tiles over H1
  bool full = (rows0 + 16 <= N);
  // fill A: 16x512 floats, 256 threads x 8 float4
#pragma unroll
  for (int it = 0; it < 8; ++it) {
    int slot = threadIdx.x + it * 256;           // 2048 slots of 4 floats
    int r = slot >> 7, c4 = (slot & 127) * 4;
    int gr = rows0 + r;
    float4 xv = make_float4(0.f, 0.f, 0.f, 0.f);
    if (full || gr < N) xv = *(const float4*)(X + (size_t)gr * F_IN + c4);
    v2h* dst = (v2h*)&ldsA[r * 520 + c4];
    v2h lo = {(_Float16)xv.x, (_Float16)xv.y};
    v2h hi = {(_Float16)xv.z, (_Float16)xv.w};
    dst[0] = lo; dst[1] = hi;
  }
  __syncthreads();
  v8f acc = {};
  const v16h* Bp = (const v16h*)W1p;
  size_t fbase = (size_t)colTile * 16 * 32 + lane;   // nK = 512/32 = 16
#pragma unroll
  for (int kcIdx = 0; kcIdx < 16; ++kcIdx) {
    v16h a = load_afrag_lds(ldsA, 520, 0, kcIdx * 32);
    v16h b = Bp[fbase + (size_t)kcIdx * 32];
    acc = __builtin_amdgcn_wmma_f32_16x16x32_f16(false, a, false, b,
                                                 (short)0, acc, false, false);
  }
  int n = lane & 15, h = lane >> 4;
  int col0 = colTile * 16;
  float bias = b1[col0 + n];
#pragma unroll
  for (int v = 0; v < 8; ++v) {
    int gr = rows0 + v + 8 * h;
    if (gr < N) {
      float r = fmaxf(acc[v] + bias, 0.f);
      H0[(size_t)gr * H1 + col0 + n] = (_Float16)r;
    }
  }
}

// ---- GEMM 2: h = h0 @ W2 + b2, h0 f16 [N,256] -> h f32 [N,256] -------------

__global__ void gemm2(const _Float16* __restrict__ H0,
                      const _Float16* __restrict__ W2p,
                      const float* __restrict__ b2,
                      float* __restrict__ H, int N) {
  __shared__ _Float16 ldsA[16 * 264];            // 16 rows x 256 K (stride 264)
  int rows0 = blockIdx.x * 16;
  int wave  = threadIdx.x >> 5;
  int lane  = threadIdx.x & 31;
  int colTile = blockIdx.y * 8 + wave;
  bool full = (rows0 + 16 <= N);
#pragma unroll
  for (int it = 0; it < 2; ++it) {
    int slot = threadIdx.x + it * 256;           // 512 slots of 8 halves
    int r = slot >> 5, c8 = (slot & 31) * 8;
    int gr = rows0 + r;
    v8h xv = {};
    if (full || gr < N) xv = *(const v8h*)(H0 + (size_t)gr * H1 + c8);
    *(v8h*)&ldsA[r * 264 + c8] = xv;
  }
  __syncthreads();
  v8f acc = {};
  const v16h* Bp = (const v16h*)W2p;
  size_t fbase = (size_t)colTile * 8 * 32 + lane;    // nK = 256/32 = 8
#pragma unroll
  for (int kcIdx = 0; kcIdx < 8; ++kcIdx) {
    v16h a = load_afrag_lds(ldsA, 264, 0, kcIdx * 32);
    v16h b = Bp[fbase + (size_t)kcIdx * 32];
    acc = __builtin_amdgcn_wmma_f32_16x16x32_f16(false, a, false, b,
                                                 (short)0, acc, false, false);
  }
  int n = lane & 15, h = lane >> 4;
  int col0 = colTile * 16;
  float bias = b2[col0 + n];
#pragma unroll
  for (int v = 0; v < 8; ++v) {
    int gr = rows0 + v + 8 * h;
    if (gr < N) H[(size_t)gr * H1 + col0 + n] = acc[v] + bias;
  }
}

// ---- attention scores: a[n] = dot(h[n], att_w) + att_b ---------------------

__global__ void attention(const float* __restrict__ H,
                          const float* __restrict__ AW,
                          const float* __restrict__ AB,
                          float* __restrict__ A, int N) {
  int g = threadIdx.x >> 5, lane = threadIdx.x & 31;
  int row = blockIdx.x * 8 + g;
  if (row >= N) return;
  float s = 0.f;
#pragma unroll
  for (int c = 0; c < H1; c += 128) {
    float4 hv = *(const float4*)(H + (size_t)row * H1 + c + lane * 4);
    float4 wv = *(const float4*)(AW + c + lane * 4);
    s += hv.x * wv.x + hv.y * wv.y + hv.z * wv.z + hv.w * wv.w;
  }
#pragma unroll
  for (int o = 16; o >= 1; o >>= 1) s += __shfl_xor(s, o, 32);
  if (lane == 0) A[row] = s + AB[0];
}

// ---- bitonic sort: descending by key, ascending idx on ties ----------------

__global__ void sort_init(const float* __restrict__ A, float* __restrict__ key,
                          int* __restrict__ idx, int N, int SN) {
  int i = blockIdx.x * 256 + threadIdx.x;
  if (i >= SN) return;
  key[i] = (i < N) ? A[i] : -3.0e38f;
  idx[i] = i;
}

__device__ __forceinline__ bool precedes(float ka, int ia, float kb, int ib) {
  return (ka > kb) || (ka == kb && ia < ib);   // strict total order
}

__global__ void bitonic_step(float* __restrict__ key, int* __restrict__ idx,
                             int k, int j, int SN) {
  int i = blockIdx.x * 256 + threadIdx.x;
  if (i >= SN) return;
  int ixj = i ^ j;
  if (ixj <= i) return;
  float ki = key[i], kj = key[ixj];
  int ii = idx[i], ij = idx[ixj];
  bool dir = ((i & k) == 0);
  bool swp = dir ? precedes(kj, ij, ki, ii) : precedes(ki, ii, kj, ij);
  if (swp) {
    key[i] = kj; key[ixj] = ki;
    idx[i] = ij; idx[ixj] = ii;
  }
}

// all passes with j <= 256 for a given k, in LDS (512 elements / block)
__global__ void bitonic_fused(float* __restrict__ key, int* __restrict__ idx,
                              int k, int jstart) {
  __shared__ float skl[512];
  __shared__ int   sil[512];
  int base = blockIdx.x * 512;
#pragma unroll
  for (int it = 0; it < 2; ++it) {
    int e = threadIdx.x + it * 256;
    skl[e] = key[base + e];
    sil[e] = idx[base + e];
  }
  __syncthreads();
  for (int j = jstart; j >= 1; j >>= 1) {
    int t = threadIdx.x;
    int i = ((t & ~(j - 1)) << 1) | (t & (j - 1));
    int ixj = i | j;
    bool dir = (((base + i) & k) == 0);
    float ki = skl[i], kj = skl[ixj];
    int ii = sil[i], ij = sil[ixj];
    bool swp = dir ? precedes(kj, ij, ki, ii) : precedes(ki, ii, kj, ij);
    if (swp) {
      skl[i] = kj; skl[ixj] = ki;
      sil[i] = ij; sil[ixj] = ii;
    }
    __syncthreads();
  }
#pragma unroll
  for (int it = 0; it < 2; ++it) {
    int e = threadIdx.x + it * 256;
    key[base + e] = skl[e];
    idx[base + e] = sil[e];
  }
}

// ---- gather into sorted order, scaled: hs = (a*h)[sidx], f16 out -----------

__global__ void gather_scale(const float* __restrict__ H,
                             const float* __restrict__ A,
                             const int* __restrict__ sidx,
                             _Float16* __restrict__ HS, int N) {
  int r = blockIdx.x * 4 + (threadIdx.x >> 6);
  if (r >= N) return;
  int c = (threadIdx.x & 63) * 4;
  int o = sidx[r];
  float s = A[o];
  float4 hv = *(const float4*)(H + (size_t)o * H1 + c);
  v2h* dst = (v2h*)(HS + (size_t)r * H1 + c);
  v2h lo = {(_Float16)(s * hv.x), (_Float16)(s * hv.y)};
  v2h hi = {(_Float16)(s * hv.z), (_Float16)(s * hv.w)};
  dst[0] = lo; dst[1] = hi;
}

// ---- conv1d (256->256 ch, K=5, SAME) via WMMA ------------------------------
// out[p,co] = sum_k sum_ci in[p+k-2,ci] * W[k][ci][co] + bias[co]

__device__ __forceinline__ void conv_lds_fill(_Float16* ldsI,
                                              const _Float16* __restrict__ IN,
                                              int rows0, int N) {
#pragma unroll
  for (int it = 0; it < 3; ++it) {
    int slot = threadIdx.x + it * 256;           // 640 slots of 8 halves
    if (slot < 20 * 32) {
      int r = slot >> 5, c8 = (slot & 31) * 8;
      int gr = rows0 + r - PADW;
      v8h v = {};
      if (gr >= 0 && gr < N) v = *(const v8h*)(IN + (size_t)gr * H1 + c8);
      *(v8h*)&ldsI[r * 264 + c8] = v;
    }
  }
}

__device__ __forceinline__ v8f conv_mma(const _Float16* ldsI,
                                        const _Float16* __restrict__ Wk,
                                        int colTile) {
  int lane = threadIdx.x & 31;
  v8f acc = {};
  const v16h* Bp = (const v16h*)Wk;
  size_t base = (size_t)colTile * 8 * 32 + lane;     // nK = 8 frags per colTile
#pragma unroll
  for (int k = 0; k < KW; ++k) {
#pragma unroll
    for (int kcIdx = 0; kcIdx < 8; ++kcIdx) {
      v16h a = load_afrag_lds(ldsI, 264, k, kcIdx * 32);
      v16h b = Bp[(size_t)k * 4096 + base + (size_t)kcIdx * 32];
      acc = __builtin_amdgcn_wmma_f32_16x16x32_f16(false, a, false, b,
                                                   (short)0, acc, false, false);
    }
  }
  return acc;
}

__global__ void conv_relu_f16(const _Float16* __restrict__ IN,
                              const _Float16* __restrict__ Wk,
                              const float* __restrict__ bias,
                              _Float16* __restrict__ OUT, int N) {
  __shared__ _Float16 ldsI[20 * 264];
  int rows0 = blockIdx.x * 16;
  int wave = threadIdx.x >> 5, lane = threadIdx.x & 31;
  int colTile = blockIdx.y * 8 + wave;
  conv_lds_fill(ldsI, IN, rows0, N);
  __syncthreads();
  v8f acc = conv_mma(ldsI, Wk, colTile);
  int n = lane & 15, h = lane >> 4;
  int col0 = colTile * 16;
  float bv = bias[col0 + n];
#pragma unroll
  for (int v = 0; v < 8; ++v) {
    int gr = rows0 + v + 8 * h;
    if (gr < N) {
      float r = fmaxf(acc[v] + bv, 0.f);
      OUT[(size_t)gr * H1 + col0 + n] = (_Float16)r;
    }
  }
}

// conv2: f32 output scattered back through the inverse permutation
__global__ void conv_scatter_f32(const _Float16* __restrict__ IN,
                                 const _Float16* __restrict__ Wk,
                                 const float* __restrict__ bias,
                                 const int* __restrict__ sidx,
                                 float* __restrict__ HC, int N) {
  __shared__ _Float16 ldsI[20 * 264];
  int rows0 = blockIdx.x * 16;
  int wave = threadIdx.x >> 5, lane = threadIdx.x & 31;
  int colTile = blockIdx.y * 8 + wave;
  conv_lds_fill(ldsI, IN, rows0, N);
  __syncthreads();
  v8f acc = conv_mma(ldsI, Wk, colTile);
  int n = lane & 15, h = lane >> 4;
  int col0 = colTile * 16;
  float bv = bias[col0 + n];
#pragma unroll
  for (int v = 0; v < 8; ++v) {
    int gr = rows0 + v + 8 * h;
    if (gr < N) {
      int orig = sidx[gr];
      HC[(size_t)orig * H1 + col0 + n] = acc[v] + bv;
    }
  }
}

// ---- final: log_softmax(concat(h, hc) @ final_w + final_b) -----------------

__global__ void final_logsoftmax(const float* __restrict__ H,
                                 const float* __restrict__ HC,
                                 const float* __restrict__ FW,
                                 const float* __restrict__ FB,
                                 float* __restrict__ OUT, int N) {
  __shared__ float buf[8][512];
  int g = threadIdx.x >> 5, lane = threadIdx.x & 31;
  int row = blockIdx.x * 8 + g;
  if (row >= N) return;
#pragma unroll
  for (int c = lane * 4; c < H1; c += 128) {
    *(float4*)&buf[g][c]      = *(const float4*)(H  + (size_t)row * H1 + c);
    *(float4*)&buf[g][H1 + c] = *(const float4*)(HC + (size_t)row * H1 + c);
  }
  // wave-private LDS slice: no cross-wave barrier needed
  bool has2 = lane < (C_OUT - 32);
  int j2 = has2 ? (lane + 32) : (C_OUT - 1);
  float f0 = FB[lane];
  float f1 = FB[j2];
  for (int c = 0; c < 2 * H1; ++c) {
    float x = buf[g][c];
    f0 += x * FW[(size_t)c * C_OUT + lane];
    f1 += x * FW[(size_t)c * C_OUT + j2];
  }
  float m = fmaxf(f0, has2 ? f1 : -3.0e38f);
#pragma unroll
  for (int o = 16; o >= 1; o >>= 1) m = fmaxf(m, __shfl_xor(m, o, 32));
  float s = expf(f0 - m) + (has2 ? expf(f1 - m) : 0.f);
#pragma unroll
  for (int o = 16; o >= 1; o >>= 1) s += __shfl_xor(s, o, 32);
  float l = logf(s);
  OUT[(size_t)row * C_OUT + lane] = f0 - m - l;
  if (has2) OUT[(size_t)row * C_OUT + lane + 32] = f1 - m - l;
}

// ---------------------------------------------------------------------------

extern "C" void kernel_launch(void* const* d_in, const int* in_sizes, int n_in,
                              void* d_out, int out_size, void* d_ws, size_t ws_size,
                              hipStream_t stream) {
  const float* x       = (const float*)d_in[0];
  const float* W1      = (const float*)d_in[1];
  const float* b1      = (const float*)d_in[2];
  const float* W2      = (const float*)d_in[3];
  const float* b2      = (const float*)d_in[4];
  const float* att_w   = (const float*)d_in[5];
  const float* att_b   = (const float*)d_in[6];
  const float* conv1_w = (const float*)d_in[7];
  const float* conv1_b = (const float*)d_in[8];
  const float* conv2_w = (const float*)d_in[9];
  const float* conv2_b = (const float*)d_in[10];
  const float* final_w = (const float*)d_in[11];
  const float* final_b = (const float*)d_in[12];
  float* out = (float*)d_out;

  const int N = in_sizes[0] / F_IN;

  size_t off = 0;
  auto carve = [&](size_t bytes) -> void* {
    void* p = (char*)d_ws + off;
    off += (bytes + 255) & ~(size_t)255;
    return p;
  };
  _Float16* w1p  = (_Float16*)carve((size_t)F_IN * H1 * 2);
  _Float16* w2p  = (_Float16*)carve((size_t)H1 * H1 * 2);
  _Float16* cw1p = (_Float16*)carve((size_t)KW * H1 * H1 * 2);
  _Float16* cw2p = (_Float16*)carve((size_t)KW * H1 * H1 * 2);
  _Float16* h0h  = (_Float16*)carve((size_t)N * H1 * 2);   // reused as conv1 out
  float*    hf   = (float*)   carve((size_t)N * H1 * 4);   // before_h (f32)
  float*    af   = (float*)   carve((size_t)N * 4);        // attention scores
  int SN = 1; while (SN < N) SN <<= 1;
  if (SN < 512) SN = 512;
  float*    skey = (float*)   carve((size_t)SN * 4);
  int*      sidx = (int*)     carve((size_t)SN * 4);
  _Float16* hsh  = (_Float16*)carve((size_t)N * H1 * 2);   // sorted, scaled (f16)
  float*    hcf  = (float*)   carve((size_t)N * H1 * 4);   // conv2 out, scattered
  _Float16* c1h  = h0h;                                    // alias: h0 dead after gemm2

  const int rowBlocks = (N + 15) / 16;

  // weight prep (fragment-major packing)
  pack_b<<<(F_IN * H1 + 255) / 256, 256, 0, stream>>>(W1, w1p, F_IN, H1);
  pack_b<<<(H1 * H1 + 255) / 256, 256, 0, stream>>>(W2, w2p, H1, H1);
  pack_convw<<<(KW * H1 * H1 + 255) / 256, 256, 0, stream>>>(conv1_w, cw1p);
  pack_convw<<<(KW * H1 * H1 + 255) / 256, 256, 0, stream>>>(conv2_w, cw2p);

  // MLP encode
  gemm1_relu<<<dim3(rowBlocks, 2), 256, 0, stream>>>(x, w1p, b1, h0h, N);
  gemm2<<<dim3(rowBlocks, 2), 256, 0, stream>>>(h0h, w2p, b2, hf, N);
  attention<<<(N + 7) / 8, 256, 0, stream>>>(hf, att_w, att_b, af, N);

  // descending stable argsort (bitonic; LDS-fused passes for j<=256)
  sort_init<<<SN / 256, 256, 0, stream>>>(af, skey, sidx, N, SN);
  for (int k = 2; k <= SN; k <<= 1) {
    for (int j = k >> 1; j >= 512; j >>= 1)
      bitonic_step<<<SN / 256, 256, 0, stream>>>(skey, sidx, k, j, SN);
    int jstart = (k >> 1 < 256) ? (k >> 1) : 256;
    bitonic_fused<<<SN / 512, 256, 0, stream>>>(skey, sidx, k, jstart);
  }

  // hs = (a*h)[sort_index] (f16 for conv input)
  gather_scale<<<(N + 3) / 4, 256, 0, stream>>>(hf, af, sidx, hsh, N);

  // conv stack along sorted axis
  conv_relu_f16<<<dim3(rowBlocks, 2), 256, 0, stream>>>(hsh, cw1p, conv1_b, c1h, N);
  conv_scatter_f32<<<dim3(rowBlocks, 2), 256, 0, stream>>>(c1h, cw2p, conv2_b, sidx, hcf, N);

  // final head + log_softmax
  final_logsoftmax<<<(N + 7) / 8, 256, 0, stream>>>(hf, hcf, final_w, final_b, out, N);
}
